// OceanDebrisGNN_6416681140847
// MI455X (gfx1250) — compile-verified, matrix operand order checked by reading, not run on hardware
//
#include <hip/hip_runtime.h>
#include <hip/hip_bf16.h>

#define NN 50000
#define EE 400000
#define DD 128
#define HH 8
#define EHD 64
#define LL 6

typedef __attribute__((ext_vector_type(16))) __bf16 v16bf;
typedef __attribute__((ext_vector_type(8)))  float  v8f;

union Frag { uint4 q[2]; v16bf v; };   // 32B: 8 VGPRs = one 16x32 bf16 A/B fragment per lane

// ---------- wave32 helpers ----------
__device__ __forceinline__ float wave_sum32(float v) {
  #pragma unroll
  for (int m = 16; m > 0; m >>= 1) v += __shfl_xor(v, m, 32);
  return v;
}
__device__ __forceinline__ float half_sum16(float v) {
  v += __shfl_xor(v, 1, 32); v += __shfl_xor(v, 2, 32);
  v += __shfl_xor(v, 4, 32); v += __shfl_xor(v, 8, 32);
  return v;
}
// monotone float<->int map so atomicMax(int) == atomicMax(float)
__device__ __forceinline__ int   enc_f(float f) { int i = __float_as_int(f); return i >= 0 ? i : (i ^ 0x7fffffff); }
__device__ __forceinline__ float dec_f(int i)   { return __int_as_float(i >= 0 ? i : (i ^ 0x7fffffff)); }

// async global->LDS b128 copy of one 16B chunk per lane (gfx1250 ASYNCcnt path)
__device__ __forceinline__ void async_copy_b128(unsigned lds_off, unsigned long long gaddr) {
  asm volatile("global_load_async_to_lds_b128 %0, %1, off" :: "v"(lds_off), "v"(gaddr) : "memory");
}
__device__ __forceinline__ void wait_async0() {
  asm volatile("s_wait_asynccnt 0x0" ::: "memory");
}
__device__ __forceinline__ unsigned lds_addr32(const void* p) {
  return (unsigned)(size_t)p;   // generic LDS addr: aperture in [63:32], offset in [31:0]
}

// ---------- weight convert + transpose to bf16 (N-major so B-fragments are contiguous) ----------
__global__ __launch_bounds__(256) void k_wconv(
    const float* __restrict__ Wl, const float* __restrict__ Wr, const float* __restrict__ We,
    __hip_bfloat16* __restrict__ WlT, __hip_bfloat16* __restrict__ WrT, __hip_bfloat16* __restrict__ WeT)
{
  int i = blockIdx.x * 256 + threadIdx.x;
  if (i < LL * DD * DD) {
    int l = i / (DD * DD), r = i % (DD * DD), n = r / DD, k = r % DD;
    WlT[i] = __float2bfloat16(Wl[l * DD * DD + k * DD + n]);
    WrT[i] = __float2bfloat16(Wr[l * DD * DD + k * DD + n]);
  }
  if (i < LL * DD * EHD) {
    int l = i / (DD * EHD), r = i % (DD * EHD), n = r / EHD, k = r % EHD;
    WeT[i] = __float2bfloat16(We[l * EHD * DD + k * DD + n]);
  }
}

// ---------- node encoder: (N,9)@(9,128) + LN + ReLU -> h (f32) and hb (bf16) ----------
__global__ __launch_bounds__(256) void k_node_encode(
    const float* __restrict__ x, const float* __restrict__ W, const float* __restrict__ b,
    const float* __restrict__ g, const float* __restrict__ beta,
    float* __restrict__ h, __hip_bfloat16* __restrict__ hb)
{
  int wave = threadIdx.x >> 5, lane = threadIdx.x & 31;
  int n = blockIdx.x * 8 + wave;
  if (n >= NN) return;
  float xv[9];
  #pragma unroll
  for (int j = 0; j < 9; ++j) xv[j] = x[n * 9 + j];
  float v[4];
  #pragma unroll
  for (int i = 0; i < 4; ++i) {
    int c = lane * 4 + i;
    float acc = b[c];
    #pragma unroll
    for (int j = 0; j < 9; ++j) acc = fmaf(xv[j], W[j * DD + c], acc);
    v[i] = acc;
  }
  float m = wave_sum32(v[0] + v[1] + v[2] + v[3]) * (1.0f / DD);
  float var = 0.f;
  #pragma unroll
  for (int i = 0; i < 4; ++i) { float d = v[i] - m; var += d * d; }
  var = wave_sum32(var) * (1.0f / DD);
  float rstd = rsqrtf(var + 1e-5f);
  #pragma unroll
  for (int i = 0; i < 4; ++i) {
    int c = lane * 4 + i;
    float y = fmaxf((v[i] - m) * rstd * g[c] + beta[c], 0.f);
    h[(size_t)n * DD + c]  = y;
    hb[(size_t)n * DD + c] = __float2bfloat16(y);
  }
}

// ---------- edge encoder: (E,3)@(3,64) + LN + ReLU -> efb (bf16) ----------
__global__ __launch_bounds__(256) void k_edge_encode(
    const float* __restrict__ ea, const float* __restrict__ W, const float* __restrict__ b,
    const float* __restrict__ g, const float* __restrict__ beta, __hip_bfloat16* __restrict__ efb)
{
  int wave = threadIdx.x >> 5, lane = threadIdx.x & 31;
  int e = blockIdx.x * 8 + wave;
  if (e >= EE) return;
  float a0 = ea[e * 3], a1 = ea[e * 3 + 1], a2 = ea[e * 3 + 2];
  float v[2];
  #pragma unroll
  for (int i = 0; i < 2; ++i) {
    int c = lane + i * 32;
    v[i] = b[c] + a0 * W[c] + a1 * W[EHD + c] + a2 * W[2 * EHD + c];
  }
  float m = wave_sum32(v[0] + v[1]) * (1.0f / EHD);
  float var = 0.f;
  #pragma unroll
  for (int i = 0; i < 2; ++i) { float d = v[i] - m; var += d * d; }
  var = wave_sum32(var) * (1.0f / EHD);
  float rstd = rsqrtf(var + 1e-5f);
  #pragma unroll
  for (int i = 0; i < 2; ++i) {
    int c = lane + i * 32;
    float y = fmaxf((v[i] - m) * rstd * g[c] + beta[c], 0.f);
    efb[(size_t)e * EHD + c] = __float2bfloat16(y);
  }
}

// ---------- per-layer init ----------
__global__ __launch_bounds__(256) void k_init(float* __restrict__ outb, float* __restrict__ s, int* __restrict__ mxi)
{
  size_t idx = (size_t)blockIdx.x * 256 + threadIdx.x;
  if (idx < (size_t)NN * DD) outb[idx] = 0.f;
  if (idx < (size_t)NN * HH) { s[idx] = 0.f; mxi[idx] = (int)0x80000000; }
}

// ---------- node GEMMs: xl = h@Wl+bl, xr = h@Wr+br via v_wmma_f32_16x16x32_bf16 ----------
// Block = 4 waves over 16 node rows; A tile (16x128 bf16, 4KB contiguous) staged to LDS via
// global_load_async_to_lds_b128; wave w owns col tiles {2w,2w+1} of BOTH outputs (4 accums,
// 16 WMMAs/wave, A fragment reused 4x per K-step).
__global__ __launch_bounds__(128) void k_gemm_node(
    const __hip_bfloat16* __restrict__ hb,
    const __hip_bfloat16* __restrict__ WlT, const __hip_bfloat16* __restrict__ WrT,
    const float* __restrict__ bl, const float* __restrict__ br,
    float* __restrict__ xl, float* __restrict__ xr)
{
  __shared__ __align__(16) __hip_bfloat16 smem[16 * DD];   // 4KB A tile
  int tid = threadIdx.x;
  int wave = tid >> 5, lane = tid & 31;
  int half = lane >> 4, rr = lane & 15;
  int row0 = blockIdx.x * 16;

  {   // stage A: 256 chunks of 16B, 2 per thread
    unsigned l0 = lds_addr32(smem) + (unsigned)tid * 16u;
    unsigned long long g0 = (unsigned long long)(size_t)(hb + (size_t)row0 * DD) + (unsigned long long)tid * 16u;
    async_copy_b128(l0,          g0);
    async_copy_b128(l0 + 2048u,  g0 + 2048ull);
    wait_async0();
  }
  __syncthreads();

  int n0 = wave * 32 + rr;      // col tile 2w
  int n1 = n0 + 16;             // col tile 2w+1
  const uint4* la  = (const uint4*)smem;                          // 16B units
  const uint4* pl0 = (const uint4*)WlT + (size_t)n0 * (DD / 8) + half;
  const uint4* pl1 = (const uint4*)WlT + (size_t)n1 * (DD / 8) + half;
  const uint4* pr0 = (const uint4*)WrT + (size_t)n0 * (DD / 8) + half;
  const uint4* pr1 = (const uint4*)WrT + (size_t)n1 * (DD / 8) + half;
  v8f aL0 = {}; v8f aL1 = {}; v8f aR0 = {}; v8f aR1 = {};
  #pragma unroll
  for (int ks = 0; ks < 4; ++ks) {           // K = 128 = 4 x 32
    Frag a, b;
    a.q[0] = la[rr * 16 + ks * 4 + half];    // row rr, K chunk per WMMA 16-bit A layout
    a.q[1] = la[rr * 16 + ks * 4 + half + 2];
    b.q[0] = pl0[ks * 4]; b.q[1] = pl0[ks * 4 + 2];
    aL0 = __builtin_amdgcn_wmma_f32_16x16x32_bf16(false, a.v, false, b.v, (short)0, aL0, false, false);
    b.q[0] = pl1[ks * 4]; b.q[1] = pl1[ks * 4 + 2];
    aL1 = __builtin_amdgcn_wmma_f32_16x16x32_bf16(false, a.v, false, b.v, (short)0, aL1, false, false);
    b.q[0] = pr0[ks * 4]; b.q[1] = pr0[ks * 4 + 2];
    aR0 = __builtin_amdgcn_wmma_f32_16x16x32_bf16(false, a.v, false, b.v, (short)0, aR0, false, false);
    b.q[0] = pr1[ks * 4]; b.q[1] = pr1[ks * 4 + 2];
    aR1 = __builtin_amdgcn_wmma_f32_16x16x32_bf16(false, a.v, false, b.v, (short)0, aR1, false, false);
  }
  float bL0 = bl[n0], bL1 = bl[n1], bR0 = br[n0], bR1 = br[n1];
  #pragma unroll
  for (int r = 0; r < 8; ++r) {              // C layout: lane&15 = N, half picks M 0-7 / 8-15
    int row = row0 + r + half * 8;
    xl[(size_t)row * DD + n0] = aL0[r] + bL0;
    xl[(size_t)row * DD + n1] = aL1[r] + bL1;
    xr[(size_t)row * DD + n0] = aR0[r] + bR0;
    xr[(size_t)row * DD + n1] = aR1[r] + bR1;
  }
}

// ---------- fused edge GEMM + attention logit + segment-max ----------
// Block = 4 waves over 16 edges; efb A tile (16x64 bf16, 2KB contiguous) staged to LDS async;
// wave w owns head tiles {2w, 2w+1}. The ee intermediate never touches memory.
__global__ __launch_bounds__(128) void k_gemm_edge_logit(
    const __hip_bfloat16* __restrict__ efb, const __hip_bfloat16* __restrict__ WeT,
    const float* __restrict__ xl, const float* __restrict__ xr,
    const float* __restrict__ att, const int* __restrict__ ei,
    float* __restrict__ logit, int* __restrict__ mxi)
{
  __shared__ __align__(16) __hip_bfloat16 smem[16 * EHD];  // 2KB A tile
  int tid = threadIdx.x;
  int wave = tid >> 5, lane = tid & 31;
  int half = lane >> 4, rr = lane & 15;
  int e0 = blockIdx.x * 16;

  {   // stage A: 128 chunks of 16B, 1 per thread
    unsigned l0 = lds_addr32(smem) + (unsigned)tid * 16u;
    unsigned long long g0 = (unsigned long long)(size_t)(efb + (size_t)e0 * EHD) + (unsigned long long)tid * 16u;
    async_copy_b128(l0, g0);
    wait_async0();
  }
  __syncthreads();

  const uint4* la = (const uint4*)smem;      // row rr occupies bytes [rr*128, rr*128+128)
  Frag a0, a1;
  a0.q[0] = la[rr * 8 + half];     a0.q[1] = la[rr * 8 + half + 2];   // K 0..31
  a1.q[0] = la[rr * 8 + half + 4]; a1.q[1] = la[rr * 8 + half + 6];   // K 32..63
  #pragma unroll
  for (int t = 0; t < 2; ++t) {
    int ht = wave * 2 + t;
    const uint4* pb = (const uint4*)WeT + (size_t)(ht * 16 + rr) * (EHD / 8) + half;
    Frag b0, b1;
    b0.q[0] = pb[0]; b0.q[1] = pb[2];
    b1.q[0] = pb[4]; b1.q[1] = pb[6];
    v8f acc = {};
    acc = __builtin_amdgcn_wmma_f32_16x16x32_bf16(false, a0.v, false, b0.v, (short)0, acc, false, false);
    acc = __builtin_amdgcn_wmma_f32_16x16x32_bf16(false, a1.v, false, b1.v, (short)0, acc, false, false);
    int cg = ht * 16 + rr;                   // global feature column (head ht, channel rr)
    float attv = att[cg];
    #pragma unroll
    for (int r = 0; r < 8; ++r) {
      int e = e0 + r + half * 8;
      int sN = ei[e], dN = ei[EE + e];
      float v = acc[r] + xl[(size_t)sN * DD + cg] + xr[(size_t)dN * DD + cg];
      v = v > 0.f ? v : 0.2f * v;            // leaky_relu(0.2)
      float p = half_sum16(v * attv);        // reduce over the 16 channels of this head
      if (rr == 0) {
        logit[(size_t)e * HH + ht] = p;
        atomicMax(&mxi[dN * HH + ht], enc_f(p));
      }
    }
  }
}

// ---------- exp(logit - max) + segment sum ----------
__global__ __launch_bounds__(256) void k_exp(
    const float* __restrict__ logit, const int* __restrict__ mxi, const int* __restrict__ ei,
    float* __restrict__ aexp, float* __restrict__ s)
{
  size_t idx = (size_t)blockIdx.x * 256 + threadIdx.x;
  if (idx >= (size_t)EE * HH) return;
  int e = (int)(idx >> 3), hh = (int)(idx & 7);
  int dN = ei[EE + e];
  float a = __expf(logit[idx] - dec_f(mxi[dN * HH + hh]));
  aexp[idx] = a;
  unsafeAtomicAdd(&s[dN * HH + hh], a);
}

// ---------- scatter: out[dst] += alpha * xl[src], float4 loads, f32 atomics ----------
__global__ __launch_bounds__(256) void k_scatter(
    const float* __restrict__ aexp, const float* __restrict__ s, const int* __restrict__ ei,
    const float* __restrict__ xl, float* __restrict__ outb)
{
  size_t idx = (size_t)blockIdx.x * 256 + threadIdx.x;
  if (idx >= (size_t)EE * 32) return;
  int e = (int)(idx >> 5), g4 = (int)(idx & 31);
  int c0 = g4 * 4, hh = c0 >> 4;
  int sN = ei[e], dN = ei[EE + e];
  float alpha = aexp[(size_t)e * HH + hh] / (s[dN * HH + hh] + 1e-16f);
  const float4 xv = *(const float4*)(xl + (size_t)sN * DD + c0);
  float* op = outb + (size_t)dN * DD + c0;
  unsafeAtomicAdd(op + 0, alpha * xv.x);
  unsafeAtomicAdd(op + 1, alpha * xv.y);
  unsafeAtomicAdd(op + 2, alpha * xv.z);
  unsafeAtomicAdd(op + 3, alpha * xv.w);
}

// ---------- residual + LN + ReLU -> h, hb ----------
__global__ __launch_bounds__(256) void k_update(
    const float* __restrict__ outb, const float* __restrict__ cb,
    const float* __restrict__ g, const float* __restrict__ bta,
    float* __restrict__ h, __hip_bfloat16* __restrict__ hb)
{
  int wave = threadIdx.x >> 5, lane = threadIdx.x & 31;
  int n = blockIdx.x * 8 + wave;
  if (n >= NN) return;
  float v[4];
  #pragma unroll
  for (int i = 0; i < 4; ++i) {
    int c = lane * 4 + i;
    v[i] = outb[(size_t)n * DD + c] + cb[c] + h[(size_t)n * DD + c];
  }
  float m = wave_sum32(v[0] + v[1] + v[2] + v[3]) * (1.0f / DD);
  float var = 0.f;
  #pragma unroll
  for (int i = 0; i < 4; ++i) { float d = v[i] - m; var += d * d; }
  var = wave_sum32(var) * (1.0f / DD);
  float rstd = rsqrtf(var + 1e-5f);
  #pragma unroll
  for (int i = 0; i < 4; ++i) {
    int c = lane * 4 + i;
    float y = fmaxf((v[i] - m) * rstd * g[c] + bta[c], 0.f);
    h[(size_t)n * DD + c]  = y;
    hb[(size_t)n * DD + c] = __float2bfloat16(y);
  }
}

// ---------- head: relu(h@W1+b1)@W2+b2 -> sigmoid ----------
__global__ __launch_bounds__(256) void k_head(
    const float* __restrict__ h, const float* __restrict__ W1, const float* __restrict__ b1,
    const float* __restrict__ W2, const float* __restrict__ b2, float* __restrict__ outp)
{
  int wave = threadIdx.x >> 5, lane = threadIdx.x & 31;
  int n = blockIdx.x * 8 + wave;
  if (n >= NN) return;
  float a0 = b1[lane], a1 = b1[lane + 32];
  for (int k = 0; k < DD; ++k) {
    float hk = h[(size_t)n * DD + k];
    a0 = fmaf(hk, W1[k * 64 + lane], a0);
    a1 = fmaf(hk, W1[k * 64 + lane + 32], a1);
  }
  a0 = fmaxf(a0, 0.f); a1 = fmaxf(a1, 0.f);
  float t = wave_sum32(a0 * W2[lane] + a1 * W2[lane + 32]);
  if (lane == 0) outp[n] = 1.f / (1.f + __expf(-(t + b2[0])));
}

extern "C" void kernel_launch(void* const* d_in, const int* in_sizes, int n_in,
                              void* d_out, int out_size, void* d_ws, size_t ws_size,
                              hipStream_t stream) {
  const float* x    = (const float*)d_in[0];
  const int*   ei   = (const int*)d_in[1];
  const float* ea   = (const float*)d_in[2];
  const float* neW  = (const float*)d_in[3];
  const float* neb  = (const float*)d_in[4];
  const float* neg  = (const float*)d_in[5];
  const float* nebt = (const float*)d_in[6];
  const float* eeW  = (const float*)d_in[7];
  const float* eeb  = (const float*)d_in[8];
  const float* eeg  = (const float*)d_in[9];
  const float* eebt = (const float*)d_in[10];
  const float* Wl   = (const float*)d_in[11];
  const float* bl   = (const float*)d_in[12];
  const float* Wr   = (const float*)d_in[13];
  const float* br   = (const float*)d_in[14];
  const float* We   = (const float*)d_in[15];
  const float* att  = (const float*)d_in[16];
  const float* cb   = (const float*)d_in[17];
  const float* lng  = (const float*)d_in[18];
  const float* lnb  = (const float*)d_in[19];
  const float* h1W  = (const float*)d_in[20];
  const float* h1b  = (const float*)d_in[21];
  const float* h2W  = (const float*)d_in[22];
  const float* h2b  = (const float*)d_in[23];

  char* wp = (char*)d_ws;
  auto alloc = [&](size_t bytes) -> void* {
    void* p = (void*)wp; wp += (bytes + 255) & ~(size_t)255; return p;
  };
  float* h     = (float*)alloc((size_t)NN * DD * 4);
  float* xl    = (float*)alloc((size_t)NN * DD * 4);
  float* xr    = (float*)alloc((size_t)NN * DD * 4);
  float* outb  = (float*)alloc((size_t)NN * DD * 4);
  float* logit = (float*)alloc((size_t)EE * HH * 4);
  float* aexp  = (float*)alloc((size_t)EE * HH * 4);
  float* s     = (float*)alloc((size_t)NN * HH * 4);
  int*   mxi   = (int*)  alloc((size_t)NN * HH * 4);
  __hip_bfloat16* hb  = (__hip_bfloat16*)alloc((size_t)NN * DD * 2);
  __hip_bfloat16* efb = (__hip_bfloat16*)alloc((size_t)EE * EHD * 2);
  __hip_bfloat16* WlT = (__hip_bfloat16*)alloc((size_t)LL * DD * DD * 2);
  __hip_bfloat16* WrT = (__hip_bfloat16*)alloc((size_t)LL * DD * DD * 2);
  __hip_bfloat16* WeT = (__hip_bfloat16*)alloc((size_t)LL * DD * EHD * 2);

  k_wconv<<<(LL * DD * DD + 255) / 256, 256, 0, stream>>>(Wl, Wr, We, WlT, WrT, WeT);
  k_node_encode<<<(NN + 7) / 8, 256, 0, stream>>>(x, neW, neb, neg, nebt, h, hb);
  k_edge_encode<<<(EE + 7) / 8, 256, 0, stream>>>(ea, eeW, eeb, eeg, eebt, efb);

  for (int l = 0; l < LL; ++l) {
    k_init<<<((size_t)NN * DD + 255) / 256, 256, 0, stream>>>(outb, s, mxi);
    k_gemm_node<<<NN / 16, 128, 0, stream>>>(hb, WlT + (size_t)l * DD * DD, WrT + (size_t)l * DD * DD,
                                             bl + l * DD, br + l * DD, xl, xr);
    k_gemm_edge_logit<<<EE / 16, 128, 0, stream>>>(efb, WeT + (size_t)l * DD * EHD, xl, xr,
                                                   att + l * DD, ei, logit, mxi);
    k_exp<<<((size_t)EE * HH + 255) / 256, 256, 0, stream>>>(logit, mxi, ei, aexp, s);
    k_scatter<<<((size_t)EE * 32 + 255) / 256, 256, 0, stream>>>(aexp, s, ei, xl, outb);
    k_update<<<(NN + 7) / 8, 256, 0, stream>>>(outb, cb + l * DD, lng + l * DD, lnb + l * DD, h, hb);
  }
  k_head<<<(NN + 7) / 8, 256, 0, stream>>>(h, h1W, h1b, h2W, h2b, (float*)d_out);
}
